// OscillatoryNeuronGroup_68401649156426
// MI455X (gfx1250) — matrix operand order repaired and checked by reading, not compiled, not run on hardware
//
#include <hip/hip_runtime.h>
#include <hip/hip_bf16.h>

#define N_OSC 8192
#define TWO_PI_C 6.28318f   /* 2 * 3.14159, matches reference literal */
#define OMEGA_C 1.0f
#define KCOUP_C 0.5f
#define DT_C 0.01f

typedef __attribute__((ext_vector_type(16))) __bf16 v16bf;
typedef __attribute__((ext_vector_type(8)))  float  v8f;

union Frag16 {
    v16bf v;
    uint4 u[2];
};

__device__ __forceinline__ unsigned short f2bf_rne(float x) {
    union { float f; unsigned int u; } c; c.f = x;
    unsigned int r = c.u + 0x7FFFu + ((c.u >> 16) & 1u);
    return (unsigned short)(r >> 16);
}

// ---------------------------------------------------------------------------
// Kernel 1: one-time fp32 -> bf16 conversion of W (4 elements / thread)
// ---------------------------------------------------------------------------
__global__ void convert_w_bf16(const float4* __restrict__ W, ushort4* __restrict__ Wb) {
    size_t i = (size_t)blockIdx.x * blockDim.x + threadIdx.x;  // float4 index
    float4 v = W[i];
    ushort4 o;
    o.x = f2bf_rne(v.x); o.y = f2bf_rne(v.y);
    o.z = f2bf_rne(v.z); o.w = f2bf_rne(v.w);
    Wb[i] = o;
}

// ---------------------------------------------------------------------------
// Kernel 2: init — copy phase into workspace, seed bf16 sin/cos buffers
// ---------------------------------------------------------------------------
__global__ void init_state(const float* __restrict__ phase_in,
                           float* __restrict__ phase,
                           unsigned short* __restrict__ s_out,
                           unsigned short* __restrict__ c_out) {
    int i = blockIdx.x * blockDim.x + threadIdx.x;
    float ph = phase_in[i];
    phase[i] = ph;
    s_out[i] = f2bf_rne(sinf(ph));
    c_out[i] = f2bf_rne(cosf(ph));
}

// ---------------------------------------------------------------------------
// Kernel 3: one Kuramoto step.
//   Block b owns rows [16b, 16b+16). 16 waves, wave w covers K slice of 512.
//   D = W_tile(16x8192) x B(8192x16), B col0 = sin(theta), col1 = cos(theta).
//   coupling_m = cos(th_m)*D[m,0] - sin(th_m)*D[m,1]
// ---------------------------------------------------------------------------
__global__ void __launch_bounds__(512)
kuramoto_step(const unsigned short* __restrict__ Wb,
              const unsigned short* __restrict__ s_in,
              const unsigned short* __restrict__ c_in,
              unsigned short* __restrict__ s_out,
              unsigned short* __restrict__ c_out,
              float* __restrict__ phase,
              const float* __restrict__ bias,
              float* __restrict__ hist_row) {
    __shared__ float red[16 * 256];

    const int tid  = threadIdx.x;
    const int wave = tid >> 5;            // 0..15 (wave32)
    const int lane = tid & 31;
    const int l16  = lane & 15;
    const bool hi  = lane >= 16;
    const int row0 = blockIdx.x << 4;     // 16 rows per block

    // A-fragment addressing: lane (l16) -> row row0+l16 of W (bf16, row-major)
    //   lo lanes: K chunks {0..7, 16..23}; hi lanes: {8..15, 24..31}
    const uint4* Arow4 = (const uint4*)(Wb + (size_t)(row0 + l16) * N_OSC);
    // B-fragment: col 0 = sin, col 1 = cos, cols 2..15 = 0 (stay zero forever)
    //   lo lanes: K chunks {0..7, 16..23}; hi lanes: {8..15, 24..31}
    const bool bvalid = (l16 < 2);
    const uint4* Bcol4 = (const uint4*)((l16 == 1) ? c_in : s_in);

    const int kbase = wave << 9;          // 512 elements per wave (16 WMMAs)
    const int qbase = kbase >> 3;         // in uint4 (8 bf16) units
    const uint4* Abase = Arow4 + qbase + (hi ? 1 : 0);
    const uint4* Bbase = Bcol4 + qbase + (hi ? 1 : 0);

    // Double-buffered software pipeline: loads for chunk i+1 are in flight
    // before the WMMA consuming chunk i issues.
    Frag16 a[2], b[2];
    b[0].u[0] = make_uint4(0u, 0u, 0u, 0u);
    b[0].u[1] = make_uint4(0u, 0u, 0u, 0u);
    b[1].u[0] = make_uint4(0u, 0u, 0u, 0u);
    b[1].u[1] = make_uint4(0u, 0u, 0u, 0u);

    a[0].u[0] = Abase[0];
    a[0].u[1] = Abase[2];
    if (bvalid) {
        b[0].u[0] = Bbase[0];
        b[0].u[1] = Bbase[2];
    }

    v8f acc = {};
#pragma unroll
    for (int it = 0; it < 16; ++it) {
        const int cur = it & 1;
        const int nxt = cur ^ 1;
        if (it < 15) {
            const int qn = (it + 1) * 4;  // 32 bf16 per chunk = 4 uint4
            a[nxt].u[0] = Abase[qn];
            a[nxt].u[1] = Abase[qn + 2];
            __builtin_prefetch((const void*)(Abase + qn + 8), 0, 3);
            if (bvalid) {
                b[nxt].u[0] = Bbase[qn];
                b[nxt].u[1] = Bbase[qn + 2];
            }
        }
        acc = __builtin_amdgcn_wmma_f32_16x16x32_bf16(
            false, a[cur].v, false, b[cur].v, (short)0, acc, false, false);
    }

    // Spill per-wave 16x16 fp32 accumulators to LDS and reduce across waves.
#pragma unroll
    for (int r = 0; r < 8; ++r)
        red[wave * 256 + r * 32 + lane] = acc[r];
    __syncthreads();

    float tsum = 0.f;
    if (tid < 256) {
#pragma unroll
        for (int w = 0; w < 16; ++w)
            tsum += red[w * 256 + tid];
    }
    __syncthreads();
    if (tid < 256)
        red[tid] = tsum;
    __syncthreads();

    // 16 threads finish the ODE step for this block's 16 oscillators.
    if (tid < 16) {
        const int m = tid;
        // D(m,n): VGPR r=m&7; lane = n (m<8) or 16+n (m>=8)
        const int base = (m & 7) * 32 + (m < 8 ? 0 : 16);
        const float WS = red[base + 0];   // (W * sin)_i
        const float WC = red[base + 1];   // (W * cos)_i
        const int i = row0 + m;
        const float ph = phase[i];
        const float s = sinf(ph);
        const float c = cosf(ph);
        const float coupling = c * WS - s * WC;
        const float d = OMEGA_C + KCOUP_C * coupling + bias[i];
        float x = ph + d * DT_C;
        float r = fmodf(x, TWO_PI_C);     // jnp.remainder: result in [0, 2pi)
        if (r < 0.f) r += TWO_PI_C;
        phase[i] = r;
        hist_row[i] = r;
        s_out[i] = f2bf_rne(sinf(r));
        c_out[i] = f2bf_rne(cosf(r));
    }
}

// ---------------------------------------------------------------------------
extern "C" void kernel_launch(void* const* d_in, const int* in_sizes, int n_in,
                              void* d_out, int out_size, void* d_ws, size_t ws_size,
                              hipStream_t stream) {
    (void)in_sizes; (void)n_in; (void)ws_size;
    const int N = N_OSC;
    const float* phase_in = (const float*)d_in[0];
    const float* W        = (const float*)d_in[1];
    const float* bias     = (const float*)d_in[2];
    float* hist = (float*)d_out;
    const int steps = out_size / N;       // steps arrives as a device scalar; derive on host

    // Workspace layout (all offsets 16B+ aligned)
    char* ws = (char*)d_ws;
    unsigned short* Wb = (unsigned short*)ws;                 // N*N*2 = 128 MiB
    size_t off = (size_t)N * N * sizeof(unsigned short);
    float* phase = (float*)(ws + off);            off += (size_t)N * sizeof(float);
    unsigned short* sA = (unsigned short*)(ws + off); off += (size_t)N * 2;
    unsigned short* cA = (unsigned short*)(ws + off); off += (size_t)N * 2;
    unsigned short* sB = (unsigned short*)(ws + off); off += (size_t)N * 2;
    unsigned short* cB = (unsigned short*)(ws + off);

    // One-time W conversion (amortized; bf16 W = 134 MB, fits 192 MB L2)
    {
        const size_t nq = (size_t)N * N / 4;              // 16M float4s
        convert_w_bf16<<<(unsigned)(nq / 256), 256, 0, stream>>>(
            (const float4*)W, (ushort4*)Wb);
    }
    init_state<<<N / 256, 256, 0, stream>>>(phase_in, phase, sA, cA);

    for (int t = 0; t < steps; ++t) {
        const unsigned short* si = (t & 1) ? sB : sA;
        const unsigned short* ci = (t & 1) ? cB : cA;
        unsigned short* so = (t & 1) ? sA : sB;
        unsigned short* co = (t & 1) ? cA : cB;
        kuramoto_step<<<N / 16, 512, 0, stream>>>(
            Wb, si, ci, so, co, phase, bias, hist + (size_t)t * N);
    }
}
